// VGT_74045236183096
// MI455X (gfx1250) — matrix operand order repaired and checked by reading, not compiled, not run on hardware
//
#include <hip/hip_runtime.h>
#include <hip/hip_bf16.h>

#define N_ 8
#define S_ 1024
#define D_ 1024
#define H_ 16
#define DH_ 64
#define FF_ 4096
#define V_ 576
#define SGENE_ 448  // S - V
#define LN_EPS 1e-5f

typedef __attribute__((ext_vector_type(16))) __bf16 bf16x16;
typedef __attribute__((ext_vector_type(8)))  __bf16 bf16x8;
typedef __attribute__((ext_vector_type(4)))  __bf16 bf16x4;
typedef __attribute__((ext_vector_type(8)))  float  f32x8;

static __device__ __forceinline__ f32x8 wmma_bf16(bf16x16 a, bf16x16 b, f32x8 c) {
  return __builtin_amdgcn_wmma_f32_16x16x32_bf16(false, a, false, b, (short)0, c, false, false);
}

// A-fragment (16x32 bf16): lane l holds row m=l%16; element e -> K = e + 8*(l/16) + (e>=8 ? 8 : 0).
// Loads two contiguous 16B chunks from a row-major row.
static __device__ __forceinline__ bf16x16 load_a_frag(const __bf16* row, int kbase, int half) {
  bf16x8 lo = *(const bf16x8*)(row + kbase + 8 * half);
  bf16x8 hi = *(const bf16x8*)(row + kbase + 16 + 8 * half);
  return __builtin_shufflevector(lo, hi, 0,1,2,3,4,5,6,7,8,9,10,11,12,13,14,15);
}
// B-fragment (32x16 bf16): lane l holds column n=l%16; element e -> K = e + 8*(l/16) + (e>=8 ? 8 : 0).
// 'nrow' points at the data for column n laid out contiguously in K (weight row / k row / vT row).
static __device__ __forceinline__ bf16x16 load_b_frag(const __bf16* nrow, int kbase, int half) {
  bf16x8 lo = *(const bf16x8*)(nrow + kbase + 8 * half);
  bf16x8 hi = *(const bf16x8*)(nrow + kbase + 16 + 8 * half);
  return __builtin_shufflevector(lo, hi, 0,1,2,3,4,5,6,7,8,9,10,11,12,13,14,15);
}

// ---------------- fp32 -> bf16 weight conversion ----------------
__global__ void k_conv_bf16(const float* __restrict__ src, __bf16* __restrict__ dst, int n) {
  int i = blockIdx.x * blockDim.x + threadIdx.x;
  if (i < n) dst[i] = (__bf16)src[i];
}

// ---------------- block reduction helper (256 threads, wave32) ----------------
static __device__ __forceinline__ void block_reduce2(float& s, float& q, float* sbuf) {
#pragma unroll
  for (int off = 16; off; off >>= 1) { s += __shfl_xor(s, off, 32); q += __shfl_xor(q, off, 32); }
  const int wv = threadIdx.x >> 5, ln = threadIdx.x & 31;
  if (ln == 0) { sbuf[wv] = s; sbuf[8 + wv] = q; }
  __syncthreads();
  if (threadIdx.x == 0) {
    float ts = 0.f, tq = 0.f;
#pragma unroll
    for (int i = 0; i < 8; ++i) { ts += sbuf[i]; tq += sbuf[8 + i]; }
    sbuf[0] = ts; sbuf[8] = tq;
  }
  __syncthreads();
  s = sbuf[0]; q = sbuf[8];
}

// ---------------- LN1: x -> bf16 normalized ----------------
__global__ __launch_bounds__(256) void k_layernorm_bf16(
    const float* __restrict__ x, const float* __restrict__ g, const float* __restrict__ b,
    __bf16* __restrict__ out) {
  __shared__ float sbuf[16];
  const size_t base = (size_t)blockIdx.x * D_;
  float4 v = ((const float4*)(x + base))[threadIdx.x];
  float s = v.x + v.y + v.z + v.w;
  float q = v.x * v.x + v.y * v.y + v.z * v.z + v.w * v.w;
  block_reduce2(s, q, sbuf);
  const float mean = s * (1.0f / D_);
  const float rstd = rsqrtf(q * (1.0f / D_) - mean * mean + LN_EPS);
  const int j = threadIdx.x * 4;
  bf16x4 o;
  o[0] = (__bf16)((v.x - mean) * rstd * g[j + 0] + b[j + 0]);
  o[1] = (__bf16)((v.y - mean) * rstd * g[j + 1] + b[j + 1]);
  o[2] = (__bf16)((v.z - mean) * rstd * g[j + 2] + b[j + 2]);
  o[3] = (__bf16)((v.w - mean) * rstd * g[j + 3] + b[j + 3]);
  *(bf16x4*)(out + base + j) = o;
}

// ---------------- QKV projection, per head. Also produces vT (transposed V). ----------------
__global__ __launch_bounds__(256) void k_qkv(
    const __bf16* __restrict__ xn,
    const __bf16* __restrict__ Wqb, const __bf16* __restrict__ Wkb, const __bf16* __restrict__ Wvb,
    const float* __restrict__ bq, const float* __restrict__ bk, const float* __restrict__ bv,
    __bf16* __restrict__ qo, __bf16* __restrict__ ko, __bf16* __restrict__ vT) {
  const int nh = blockIdx.x;           // n*H + h
  const int n = nh / H_, h = nh % H_;
  const int wv = threadIdx.x >> 5, lane = threadIdx.x & 31;
  const int half = lane >> 4, lc = lane & 15;
  const int s0 = blockIdx.y * 128 + wv * 16;

  const __bf16* xrow = xn + ((size_t)n * S_ + s0 + lc) * D_ + h * DH_;
  const bf16x16 a0 = load_a_frag(xrow, 0, half);
  const bf16x16 a1 = load_a_frag(xrow, 32, half);

  const __bf16* Ws[3] = {Wqb, Wkb, Wvb};
  const float*  bs[3] = {bq, bk, bv};
#pragma unroll
  for (int m = 0; m < 3; ++m) {
    const __bf16* Wh = Ws[m] + (size_t)h * DH_ * DH_;
#pragma unroll
    for (int nt = 0; nt < 4; ++nt) {
      const __bf16* wrow = Wh + (size_t)(nt * 16 + lc) * DH_;  // out = x @ W^T: B col n = W row n
      f32x8 c = {};
      c = wmma_bf16(a0, load_b_frag(wrow, 0, half), c);
      c = wmma_bf16(a1, load_b_frag(wrow, 32, half), c);
      const int e = nt * 16 + lc;
      const float bias = bs[m][h * DH_ + e];
#pragma unroll
      for (int r = 0; r < 8; ++r) {
        const float val = c[r] + bias;
        const int mrow = s0 + r + 8 * half;
        if (m == 0)      qo[((size_t)nh * S_ + mrow) * DH_ + e] = (__bf16)val;
        else if (m == 1) ko[((size_t)nh * S_ + mrow) * DH_ + e] = (__bf16)val;
        else             vT[((size_t)nh * DH_ + e) * S_ + mrow] = (__bf16)val;
      }
    }
  }
}

// ---------------- sum of guided weights per batch ----------------
__global__ void k_gsum(const float* __restrict__ g, float* __restrict__ gsum) {
  __shared__ float sb[8];
  const int n = blockIdx.x;
  float s = 0.f;
  for (int j = threadIdx.x; j < V_; j += 256) s += g[n * V_ + j];
#pragma unroll
  for (int off = 16; off; off >>= 1) s += __shfl_xor(s, off, 32);
  if ((threadIdx.x & 31) == 0) sb[threadIdx.x >> 5] = s;
  __syncthreads();
  if (threadIdx.x == 0) {
    float t = 0.f;
    for (int i = 0; i < 8; ++i) t += sb[i];
    gsum[n] = t;
  }
}

// ---------------- gv[n,h,e] = sum_j g[n,j] * v[n,h,448+j,e] (rank-1 vision term) ----------------
__global__ void k_gv(const float* __restrict__ g, const __bf16* __restrict__ vT,
                     float* __restrict__ gv) {
  const int nh = blockIdx.x, n = nh / H_;
  const int e = threadIdx.x;  // 64 threads
  const __bf16* vrow = vT + ((size_t)nh * DH_ + e) * S_ + SGENE_;
  const float* gr = g + n * V_;
  float acc = 0.f;
  for (int j = 0; j < V_; ++j) acc += gr[j] * (float)vrow[j];
  gv[nh * DH_ + e] = acc;
}

// ---------------- attention: scores + softmax + guided renorm + att write + att@V ----------------
__global__ __launch_bounds__(256) void k_attn(
    const __bf16* __restrict__ q, const __bf16* __restrict__ kk, const __bf16* __restrict__ vT,
    const float* __restrict__ g, const float* __restrict__ gsum, const float* __restrict__ gv,
    float* __restrict__ att, float* __restrict__ ao) {
  __shared__ __attribute__((aligned(32))) __bf16 ldsA[8][32][16];  // per-wave D->A bounce tile
  const int nh = blockIdx.x, n = nh / H_, h = nh % H_;
  const int wv = threadIdx.x >> 5, lane = threadIdx.x & 31;
  const int half = lane >> 4, lc = lane & 15;
  const int s0 = blockIdx.y * 128 + wv * 16;

  const __bf16* qrow = q + ((size_t)nh * S_ + s0 + lc) * DH_;
  const bf16x16 a0 = load_a_frag(qrow, 0, half);   // kept in VGPRs for the whole kernel
  const bf16x16 a1 = load_a_frag(qrow, 32, half);
  const __bf16* kbase = kk + (size_t)nh * S_ * DH_;

  // ---- pass 1: online (max, gene-sum, vis-sum) over all 1024 columns ----
  float rm[8], sg[8], sv[8];
#pragma unroll
  for (int r = 0; r < 8; ++r) { rm[r] = -1e30f; sg[r] = 0.f; sv[r] = 0.f; }
  for (int t0 = 0; t0 < S_; t0 += 16) {
    const __bf16* krow = kbase + (size_t)(t0 + lc) * DH_;  // scores B: column t = k row t
    f32x8 c = {};
    c = wmma_bf16(a0, load_b_frag(krow, 0, half), c);
    c = wmma_bf16(a1, load_b_frag(krow, 32, half), c);
    const bool isVis = (t0 >= SGENE_);
#pragma unroll
    for (int r = 0; r < 8; ++r) {
      const float xv = c[r] * 0.125f;  // 1/sqrt(DH)
      const float nm = fmaxf(rm[r], xv);
      const float corr = __expf(rm[r] - nm);
      const float e = __expf(xv - nm);
      sg[r] = sg[r] * corr + (isVis ? 0.f : e);
      sv[r] = sv[r] * corr + (isVis ? e : 0.f);
      rm[r] = nm;
    }
  }
  // merge across the 16 lanes sharing each row (xor<16 keeps lane-halves intact)
#pragma unroll
  for (int off = 8; off; off >>= 1) {
#pragma unroll
    for (int r = 0; r < 8; ++r) {
      const float om = __shfl_xor(rm[r], off, 32);
      const float og = __shfl_xor(sg[r], off, 32);
      const float ov = __shfl_xor(sv[r], off, 32);
      const float nm = fmaxf(rm[r], om);
      const float c1 = __expf(rm[r] - nm), c2 = __expf(om - nm);
      sg[r] = sg[r] * c1 + og * c2;
      sv[r] = sv[r] * c1 + ov * c2;
      rm[r] = nm;
    }
  }
  const float gsumN = gsum[n];
  float inv[8], scl[8];
#pragma unroll
  for (int r = 0; r < 8; ++r) {
    inv[r] = 1.0f / (sg[r] + sv[r]);          // 1/denominator
    scl[r] = sv[r] * inv[r] / gsumN;          // vis_mass / g.sum()
  }

  // ---- pass 2: recompute gene tiles, write att, fuse att@V via LDS bounce ----
  float* attb = att + ((size_t)nh * S_ + s0) * S_;
  f32x8 acc[4] = {{}, {}, {}, {}};
  for (int t0 = 0; t0 < SGENE_; t0 += 32) {
#pragma unroll
    for (int tt = 0; tt < 2; ++tt) {
      const int tb = t0 + tt * 16;
      const __bf16* krow = kbase + (size_t)(tb + lc) * DH_;
      f32x8 c = {};
      c = wmma_bf16(a0, load_b_frag(krow, 0, half), c);
      c = wmma_bf16(a1, load_b_frag(krow, 32, half), c);
#pragma unroll
      for (int r = 0; r < 8; ++r) {
        const float p = __expf(c[r] * 0.125f - rm[r]) * inv[r];
        const int mrow = r + 8 * half;
        attb[(size_t)mrow * S_ + tb + lc] = p;
        const int kidx = tt * 16 + lc;  // (m,k) -> A-frag lane/elem
        ldsA[wv][mrow + 16 * ((kidx >> 3) & 1)][(kidx & 7) + 8 * (kidx >> 4)] = (__bf16)p;
      }
    }
    asm volatile("s_wait_dscnt 0" ::: "memory");  // cross-lane LDS bounce, same wave (DS in-order)
    const bf16x16 af = *(const bf16x16*)(&ldsA[wv][lane][0]);
#pragma unroll
    for (int nt = 0; nt < 4; ++nt) {
      const __bf16* vrow = vT + ((size_t)nh * DH_ + nt * 16 + lc) * S_;  // att@V B: col e = vT row e
      acc[nt] = wmma_bf16(af, load_b_frag(vrow, t0, half), acc[nt]);
    }
  }
  // vision columns of att: g[j] * scale[row]
  const float* gr = g + n * V_;
  for (int t0 = SGENE_; t0 < S_; t0 += 16) {
    const float gval = gr[t0 - SGENE_ + lc];
#pragma unroll
    for (int r = 0; r < 8; ++r)
      attb[(size_t)(r + 8 * half) * S_ + t0 + lc] = gval * scl[r];
  }
  // ao = acc + scale[row] * gv[e]
#pragma unroll
  for (int nt = 0; nt < 4; ++nt) {
    const int e = nt * 16 + lc;
    const float gve = gv[nh * DH_ + e];
#pragma unroll
    for (int r = 0; r < 8; ++r) {
      const int mrow = s0 + r + 8 * half;
      ao[((size_t)n * S_ + mrow) * D_ + h * DH_ + e] = acc[nt][r] + scl[r] * gve;
    }
  }
}

// ---------------- residual + LN2: out = x + ao -> d_out; yn = LN2(out) bf16 ----------------
__global__ __launch_bounds__(256) void k_resid_ln2(
    const float* __restrict__ x, const float* __restrict__ ao,
    const float* __restrict__ g, const float* __restrict__ b,
    float* __restrict__ outres, __bf16* __restrict__ yn) {
  __shared__ float sbuf[16];
  const size_t base = (size_t)blockIdx.x * D_;
  float4 xv = ((const float4*)(x + base))[threadIdx.x];
  float4 av = ((const float4*)(ao + base))[threadIdx.x];
  float4 o;
  o.x = xv.x + av.x; o.y = xv.y + av.y; o.z = xv.z + av.z; o.w = xv.w + av.w;
  ((float4*)(outres + base))[threadIdx.x] = o;
  float s = o.x + o.y + o.z + o.w;
  float q = o.x * o.x + o.y * o.y + o.z * o.z + o.w * o.w;
  block_reduce2(s, q, sbuf);
  const float mean = s * (1.0f / D_);
  const float rstd = rsqrtf(q * (1.0f / D_) - mean * mean + LN_EPS);
  const int j = threadIdx.x * 4;
  bf16x4 yo;
  yo[0] = (__bf16)((o.x - mean) * rstd * g[j + 0] + b[j + 0]);
  yo[1] = (__bf16)((o.y - mean) * rstd * g[j + 1] + b[j + 1]);
  yo[2] = (__bf16)((o.z - mean) * rstd * g[j + 2] + b[j + 2]);
  yo[3] = (__bf16)((o.w - mean) * rstd * g[j + 3] + b[j + 3]);
  *(bf16x4*)(yn + base + j) = yo;
}

// ---------------- FFN1: hmid = gelu(yn @ W1^T + b1), bf16 out ----------------
// 2 M-tiles x 4 N-tiles per wave (32x64 output): 8 WMMAs per 12 b128 loads.
__global__ __launch_bounds__(256) void k_ffn1(
    const __bf16* __restrict__ yn, const __bf16* __restrict__ W1b, const float* __restrict__ b1,
    __bf16* __restrict__ hmid) {
  const int wv = threadIdx.x >> 5, lane = threadIdx.x & 31;
  const int half = lane >> 4, lc = lane & 15;
  const int m0 = blockIdx.x * 256 + wv * 32;
  const int n0 = blockIdx.y * 64;
  const __bf16* arow0 = yn + (size_t)(m0 + lc) * D_;
  const __bf16* arow1 = yn + (size_t)(m0 + 16 + lc) * D_;
  const __bf16* brow[4];
#pragma unroll
  for (int nt = 0; nt < 4; ++nt) brow[nt] = W1b + (size_t)(n0 + nt * 16 + lc) * D_;
  f32x8 acc[2][4] = {};
  for (int k0 = 0; k0 < D_; k0 += 32) {
    __builtin_prefetch(arow0 + k0 + 128, 0, 0);  // global_prefetch_b8
    const bf16x16 a0 = load_a_frag(arow0, k0, half);
    const bf16x16 a1 = load_a_frag(arow1, k0, half);
#pragma unroll
    for (int nt = 0; nt < 4; ++nt) {
      const bf16x16 bb = load_b_frag(brow[nt], k0, half);
      acc[0][nt] = wmma_bf16(a0, bb, acc[0][nt]);
      acc[1][nt] = wmma_bf16(a1, bb, acc[1][nt]);
    }
  }
#pragma unroll
  for (int mt = 0; mt < 2; ++mt) {
#pragma unroll
    for (int nt = 0; nt < 4; ++nt) {
      const int e = n0 + nt * 16 + lc;
      const float bias = b1[e];
#pragma unroll
      for (int r = 0; r < 8; ++r) {
        const float xg = acc[mt][nt][r] + bias;
        const float gl = 0.5f * xg * (1.0f + erff(xg * 0.70710678f));  // exact gelu
        hmid[(size_t)(m0 + mt * 16 + r + 8 * half) * FF_ + e] = (__bf16)gl;
      }
    }
  }
}

// ---------------- FFN2: d_out += hmid @ W2^T + b2 ----------------
__global__ __launch_bounds__(256) void k_ffn2(
    const __bf16* __restrict__ hmid, const __bf16* __restrict__ W2b, const float* __restrict__ b2,
    float* __restrict__ outres) {
  const int wv = threadIdx.x >> 5, lane = threadIdx.x & 31;
  const int half = lane >> 4, lc = lane & 15;
  const int m0 = blockIdx.x * 256 + wv * 32;
  const int n0 = blockIdx.y * 64;
  const __bf16* arow0 = hmid + (size_t)(m0 + lc) * FF_;
  const __bf16* arow1 = hmid + (size_t)(m0 + 16 + lc) * FF_;
  const __bf16* brow[4];
#pragma unroll
  for (int nt = 0; nt < 4; ++nt) brow[nt] = W2b + (size_t)(n0 + nt * 16 + lc) * FF_;
  f32x8 acc[2][4] = {};
  for (int k0 = 0; k0 < FF_; k0 += 32) {
    __builtin_prefetch(arow0 + k0 + 128, 0, 0);
    const bf16x16 a0 = load_a_frag(arow0, k0, half);
    const bf16x16 a1 = load_a_frag(arow1, k0, half);
#pragma unroll
    for (int nt = 0; nt < 4; ++nt) {
      const bf16x16 bb = load_b_frag(brow[nt], k0, half);
      acc[0][nt] = wmma_bf16(a0, bb, acc[0][nt]);
      acc[1][nt] = wmma_bf16(a1, bb, acc[1][nt]);
    }
  }
#pragma unroll
  for (int mt = 0; mt < 2; ++mt) {
#pragma unroll
    for (int nt = 0; nt < 4; ++nt) {
      const int e = n0 + nt * 16 + lc;
      const float bias = b2[e];
#pragma unroll
      for (int r = 0; r < 8; ++r) {
        const size_t idx = (size_t)(m0 + mt * 16 + r + 8 * half) * D_ + e;
        outres[idx] = outres[idx] + acc[mt][nt][r] + bias;  // final = out + y
      }
    }
  }
}

extern "C" void kernel_launch(void* const* d_in, const int* in_sizes, int n_in,
                              void* d_out, int out_size, void* d_ws, size_t ws_size,
                              hipStream_t stream) {
  (void)in_sizes; (void)n_in; (void)out_size; (void)ws_size;
  const float* x    = (const float*)d_in[0];
  const float* g    = (const float*)d_in[1];
  const float* ln1g = (const float*)d_in[2];
  const float* ln1b = (const float*)d_in[3];
  const float* Wq   = (const float*)d_in[4];
  const float* bq   = (const float*)d_in[5];
  const float* Wk   = (const float*)d_in[6];
  const float* bk   = (const float*)d_in[7];
  const float* Wv   = (const float*)d_in[8];
  const float* bv   = (const float*)d_in[9];
  const float* ln2g = (const float*)d_in[10];
  const float* ln2b = (const float*)d_in[11];
  const float* W1   = (const float*)d_in[12];
  const float* b1   = (const float*)d_in[13];
  const float* W2   = (const float*)d_in[14];
  const float* b2   = (const float*)d_in[15];

  char* ws = (char*)d_ws;
  size_t off = 0;
  auto alloc = [&](size_t bytes) -> char* {
    char* p = ws + off;
    off += (bytes + 255) & ~(size_t)255;
    return p;
  };
  // Region A (exactly 64MB): xnb/q/k/vT; dead after k_attn, aliased by hmid (64MB).
  __bf16* xnb = (__bf16*)alloc((size_t)N_ * S_ * D_ * 2);
  __bf16* qb  = (__bf16*)alloc((size_t)N_ * H_ * S_ * DH_ * 2);
  __bf16* kb  = (__bf16*)alloc((size_t)N_ * H_ * S_ * DH_ * 2);
  __bf16* vTb = (__bf16*)alloc((size_t)N_ * H_ * DH_ * S_ * 2);
  __bf16* hmid = (__bf16*)ws;  // aliases region A (N*S*FF bf16 = 64MB)
  __bf16* Wqb = (__bf16*)alloc((size_t)H_ * DH_ * DH_ * 2);
  __bf16* Wkb = (__bf16*)alloc((size_t)H_ * DH_ * DH_ * 2);
  __bf16* Wvb = (__bf16*)alloc((size_t)H_ * DH_ * DH_ * 2);
  __bf16* W1b = (__bf16*)alloc((size_t)FF_ * D_ * 2);
  __bf16* W2b = (__bf16*)alloc((size_t)D_ * FF_ * 2);
  __bf16* ynb = (__bf16*)alloc((size_t)N_ * S_ * D_ * 2);
  float*  ao  = (float*)alloc((size_t)N_ * S_ * D_ * 4);
  float*  gv  = (float*)alloc((size_t)N_ * H_ * DH_ * 4);
  float*  gsum = (float*)alloc((size_t)N_ * 4);

  float* outres = (float*)d_out;                       // out + y
  float* att = (float*)d_out + (size_t)N_ * S_ * D_;   // attention matrix

  const dim3 b256(256);
  k_conv_bf16<<<dim3((H_ * DH_ * DH_ + 255) / 256), b256, 0, stream>>>(Wq, Wqb, H_ * DH_ * DH_);
  k_conv_bf16<<<dim3((H_ * DH_ * DH_ + 255) / 256), b256, 0, stream>>>(Wk, Wkb, H_ * DH_ * DH_);
  k_conv_bf16<<<dim3((H_ * DH_ * DH_ + 255) / 256), b256, 0, stream>>>(Wv, Wvb, H_ * DH_ * DH_);
  k_conv_bf16<<<dim3((FF_ * D_ + 255) / 256), b256, 0, stream>>>(W1, W1b, FF_ * D_);
  k_conv_bf16<<<dim3((D_ * FF_ + 255) / 256), b256, 0, stream>>>(W2, W2b, D_ * FF_);

  k_layernorm_bf16<<<dim3(N_ * S_), b256, 0, stream>>>(x, ln1g, ln1b, xnb);
  k_qkv<<<dim3(N_ * H_, S_ / 128), b256, 0, stream>>>(xnb, Wqb, Wkb, Wvb, bq, bk, bv, qb, kb, vTb);
  k_gsum<<<dim3(N_), b256, 0, stream>>>(g, gsum);
  k_gv<<<dim3(N_ * H_), dim3(64), 0, stream>>>(g, vTb, gv);
  k_attn<<<dim3(N_ * H_, S_ / 128), b256, 0, stream>>>(qb, kb, vTb, g, gsum, gv, att, ao);
  k_resid_ln2<<<dim3(N_ * S_), b256, 0, stream>>>(x, ao, ln2g, ln2b, outres, ynb);
  k_ffn1<<<dim3(N_ * S_ / 256, FF_ / 64), b256, 0, stream>>>(ynb, W1b, b1, hmid);
  k_ffn2<<<dim3(N_ * S_ / 256, D_ / 64), b256, 0, stream>>>(hmid, W2b, b2, outres);
}
